// MultiHeadAttention_25975962206821
// MI455X (gfx1250) — compile-verified, matrix operand order checked by reading, not run on hardware
//
#include <hip/hip_runtime.h>
#include <hip/hip_bf16.h>
#include <math.h>

// ---------------- problem constants ----------------
constexpr int DMODEL = 512;
constexpr int NH     = 8;
constexpr int DHEAD  = 64;
constexpr int BB     = 4;
constexpr int SEQ    = 2048;
constexpr int MROWS  = BB * SEQ;          // 8192 rows for all row-major GEMMs

typedef __attribute__((ext_vector_type(16))) _Float16 v16h;
typedef __attribute__((ext_vector_type(8)))  _Float16 v8h;
typedef __attribute__((ext_vector_type(8)))  float    v8f;
typedef __attribute__((ext_vector_type(4)))  float    v4f;

// ---------------------------------------------------------------------------
// Fragment loaders for V_WMMA_F32_16X16X32_F16 (wave32).
// A-matrix 16x32 f16 layout (ISA 7.12.2): lane r = outer index (M for A, N for B),
// lanes 0-15 hold k = {0..7, 16..23}, lanes 16-31 hold k = {8..15, 24..31},
// packed 2 halves per VGPR.  B uses the symmetric layout, so loading an
// N x K row-major matrix with this loader yields B(k,n) = src[n][k], i.e.
// C = A * B^T for two row-major operands.
// ---------------------------------------------------------------------------
__device__ __forceinline__ v16h frag_from_h(const _Float16* p, int ld, int lane) {
    int r  = lane & 15;
    int kb = (lane & 16) ? 8 : 0;
    const _Float16* row = p + (size_t)r * ld + kb;
    v8h lo = *(const v8h*)(row);
    v8h hi = *(const v8h*)(row + 16);
    v16h f;
#pragma unroll
    for (int i = 0; i < 8; ++i) { f[i] = lo[i]; f[i + 8] = hi[i]; }
    return f;
}

__device__ __forceinline__ v16h frag_from_f32(const float* p, int ld, int lane) {
    int r  = lane & 15;
    int kb = (lane & 16) ? 8 : 0;
    const float* row = p + (size_t)r * ld + kb;
    v4f a0 = *(const v4f*)(row);
    v4f a1 = *(const v4f*)(row + 4);
    v4f b0 = *(const v4f*)(row + 16);
    v4f b1 = *(const v4f*)(row + 20);
    v16h f;
#pragma unroll
    for (int i = 0; i < 4; ++i) {
        f[i]      = (_Float16)a0[i];
        f[i + 4]  = (_Float16)a1[i];
        f[i + 8]  = (_Float16)b0[i];
        f[i + 12] = (_Float16)b1[i];
    }
    return f;
}

// C/D fragment (v8f): lanes 0-15 -> (M=j, N=lane); lanes 16-31 -> (M=8+j, N=lane-16)

// ---------------------------------------------------------------------------
// Weight transpose + fp32 -> fp16 convert:  Wt[n*512 + k] = W[k*512 + n]
// ---------------------------------------------------------------------------
__global__ __launch_bounds__(256) void wtrans_kernel(const float* __restrict__ W,
                                                     _Float16* __restrict__ Wt) {
    int idx = blockIdx.x * 256 + threadIdx.x;      // 0 .. 512*512-1
    int n = idx >> 9;
    int k = idx & 511;
    Wt[idx] = (_Float16)W[(size_t)k * DMODEL + n];
}

// ---------------------------------------------------------------------------
// GEMM: out[M,512] = X[M,512] (fp32) @ Wt^T + bias, WMMA f16->f32.
//   mode 0: store f16 row-major (Q/K projections)
//   mode 1: store f16 transposed per head: v_t[(b*NH+h)*64 + d][s]  (V proj)
//   mode 2: store f32 row-major (final output projection)
// grid = (M/128, 512/64), block = 256 (8 waves); wave w -> 16-row tile.
// ---------------------------------------------------------------------------
__global__ __launch_bounds__(256) void gemm_kernel(const float* __restrict__ X,
                                                   const _Float16* __restrict__ Wt,
                                                   const float* __restrict__ bias,
                                                   void* __restrict__ out, int mode) {
    int wave = threadIdx.x >> 5;
    int lane = threadIdx.x & 31;
    int m0 = blockIdx.x * 128 + wave * 16;
    int n0 = blockIdx.y * 64;

    v8f acc[4] = {v8f{0.f}, v8f{0.f}, v8f{0.f}, v8f{0.f}};
#pragma unroll 4
    for (int k0 = 0; k0 < DMODEL; k0 += 32) {
        v16h a = frag_from_f32(X + (size_t)m0 * DMODEL + k0, DMODEL, lane);
#pragma unroll
        for (int t = 0; t < 4; ++t) {
            v16h b = frag_from_h(Wt + (size_t)(n0 + 16 * t) * DMODEL + k0, DMODEL, lane);
            acc[t] = __builtin_amdgcn_wmma_f32_16x16x32_f16(false, a, false, b,
                                                            (short)0, acc[t], false, false);
        }
    }

    int n_in = lane & 15;
    int mloc = (lane & 16) ? 8 : 0;
#pragma unroll
    for (int t = 0; t < 4; ++t) {
        int n = n0 + 16 * t + n_in;
        float bv = bias[n];
#pragma unroll
        for (int j = 0; j < 8; ++j) {
            float v = acc[t][j] + bv;
            int m = m0 + mloc + j;
            if (mode == 0) {
                ((_Float16*)out)[(size_t)m * DMODEL + n] = (_Float16)v;
            } else if (mode == 1) {
                int b = m >> 11, s = m & (SEQ - 1);
                int h = n >> 6,  d = n & (DHEAD - 1);
                ((_Float16*)out)[((size_t)(b * NH + h) * DHEAD + d) * SEQ + s] = (_Float16)v;
            } else {
                ((float*)out)[(size_t)m * DMODEL + n] = v;
            }
        }
    }
}

// ---------------------------------------------------------------------------
// Fused attention: one block per (b, h, 16-row q tile).  block = 256 (8 waves).
//   stage 1: raw scores (WMMA q·k^T) * 1/sqrt(512) -> LDS [16][2048] f32
//   stage 2: mask (coalesced b32 loads) + row softmax, exp computed once;
//            normalized weights -> LDS (in place) + global via float4 stores
//   stage 3: PV via WMMA (f16 P fragments rebuilt from LDS, v_t B fragments),
//            cross-wave reduce in LDS, float4 fp32 context store.
// ---------------------------------------------------------------------------
__global__ __launch_bounds__(256) void attn_kernel(const _Float16* __restrict__ q_h,
                                                   const _Float16* __restrict__ k_h,
                                                   const _Float16* __restrict__ v_t,
                                                   const unsigned char* __restrict__ mask,
                                                   float* __restrict__ weight_out,
                                                   float* __restrict__ attn_out) {
    __shared__ float smem[16 * SEQ];     // 128 KB score / prob tile
    __shared__ float scratch[256];       // row reductions

    int bid = blockIdx.x;                // b*NH*128 + h*128 + qt
    int qt = bid & 127;
    int h  = (bid >> 7) & (NH - 1);
    int b  = bid >> 10;
    int q0 = qt * 16;

    int wave = threadIdx.x >> 5;
    int lane = threadIdx.x & 31;
    int n_in = lane & 15;
    int mloc = (lane & 16) ? 8 : 0;
    const float scale = 0.044194173824159216f;   // 1/sqrt(512)

    // ---- stage 1: raw scaled scores ----
    const _Float16* qbase = q_h + ((size_t)(b * SEQ + q0)) * DMODEL + h * DHEAD;
    v16h qa0 = frag_from_h(qbase,      DMODEL, lane);
    v16h qa1 = frag_from_h(qbase + 32, DMODEL, lane);

    for (int kt = wave; kt < 128; kt += 8) {
        int k0 = kt * 16;
        const _Float16* kbase = k_h + ((size_t)(b * SEQ + k0)) * DMODEL + h * DHEAD;
        v16h kb0 = frag_from_h(kbase,      DMODEL, lane);
        v16h kb1 = frag_from_h(kbase + 32, DMODEL, lane);
        v8f c = v8f{0.f};
        c = __builtin_amdgcn_wmma_f32_16x16x32_f16(false, qa0, false, kb0, (short)0, c, false, false);
        c = __builtin_amdgcn_wmma_f32_16x16x32_f16(false, qa1, false, kb1, (short)0, c, false, false);
#pragma unroll
        for (int j = 0; j < 8; ++j) {
            smem[(mloc + j) * SEQ + k0 + n_in] = c[j] * scale;
        }
    }
    __syncthreads();

    // ---- stage 2: mask + softmax.  thread (r, cc) owns kk = j*64 + cc*4 ----
    int r  = threadIdx.x >> 4;           // row 0..15
    int cc = threadIdx.x & 15;           // 16 threads per row, 4-elem chunks
    const unsigned char* mrow = mask + ((size_t)b * SEQ + (q0 + r)) * SEQ;
    float* srow = smem + r * SEQ;

    // pass 1: apply mask (coalesced 4-byte loads), track row max
    float lmax = -INFINITY;
#pragma unroll 4
    for (int j = 0; j < SEQ / 64; ++j) {
        int kk = j * 64 + cc * 4;
        v4f s = *(v4f*)(srow + kk);
        unsigned int mw = *(const unsigned int*)(mrow + kk);
#pragma unroll
        for (int i = 0; i < 4; ++i)
            if ((mw >> (8 * i)) & 0xffu) s[i] = -INFINITY;
        *(v4f*)(srow + kk) = s;
        lmax = fmaxf(lmax, fmaxf(fmaxf(s[0], s[1]), fmaxf(s[2], s[3])));
    }
    scratch[threadIdx.x] = lmax;
    __syncthreads();
    float rowmax = -INFINITY;
#pragma unroll
    for (int i = 0; i < 16; ++i) rowmax = fmaxf(rowmax, scratch[r * 16 + i]);
    __syncthreads();

    // pass 2: exp once, store back, accumulate row sum
    float lsum = 0.f;
#pragma unroll 4
    for (int j = 0; j < SEQ / 64; ++j) {
        int kk = j * 64 + cc * 4;
        v4f s = *(v4f*)(srow + kk);
        v4f e;
#pragma unroll
        for (int i = 0; i < 4; ++i) e[i] = __expf(s[i] - rowmax);
        *(v4f*)(srow + kk) = e;
        lsum += e[0] + e[1] + e[2] + e[3];
    }
    scratch[threadIdx.x] = lsum;
    __syncthreads();
    float rowsum = 0.f;
#pragma unroll
    for (int i = 0; i < 16; ++i) rowsum += scratch[r * 16 + i];
    float inv = 1.0f / rowsum;

    // pass 3: normalize; float4 stores to the 512MB weight output
    float* wrow = weight_out + (((size_t)(b * NH + h) * SEQ) + q0 + r) * SEQ;
#pragma unroll 4
    for (int j = 0; j < SEQ / 64; ++j) {
        int kk = j * 64 + cc * 4;
        v4f e = *(v4f*)(srow + kk);
        v4f w = e * inv;
        *(v4f*)(srow + kk) = w;      // probs for PV
        *(v4f*)(wrow + kk) = w;      // mandatory global output (b128 store)
    }
    __syncthreads();

    // ---- stage 3: out = P @ V ----
    v8f acc[4] = {v8f{0.f}, v8f{0.f}, v8f{0.f}, v8f{0.f}};
    int rr = lane & 15;
    int kb = (lane & 16) ? 8 : 0;
    for (int kt = wave; kt < 128; kt += 8) {
        int k0 = kt * 16;
        // P fragment from LDS floats -> f16 (vector LDS reads)
        const float* prow = smem + rr * SEQ + k0 + kb;
        v4f p0 = *(const v4f*)(prow);
        v4f p1 = *(const v4f*)(prow + 4);
        v4f p2 = *(const v4f*)(prow + 16);
        v4f p3 = *(const v4f*)(prow + 20);
        v16h a;
#pragma unroll
        for (int i = 0; i < 4; ++i) {
            a[i]      = (_Float16)p0[i];
            a[i + 4]  = (_Float16)p1[i];
            a[i + 8]  = (_Float16)p2[i];
            a[i + 12] = (_Float16)p3[i];
        }
#pragma unroll
        for (int t = 0; t < 4; ++t) {
            v16h bf = frag_from_h(v_t + ((size_t)(b * NH + h) * DHEAD + t * 16) * SEQ + k0,
                                  SEQ, lane);
            acc[t] = __builtin_amdgcn_wmma_f32_16x16x32_f16(false, a, false, bf,
                                                            (short)0, acc[t], false, false);
        }
    }
    __syncthreads();   // all waves done reading smem probs

    // cross-wave reduction through LDS (reuse smem: 8*16*64 floats = 32 KB)
#pragma unroll
    for (int t = 0; t < 4; ++t)
#pragma unroll
        for (int j = 0; j < 8; ++j)
            smem[wave * 1024 + (mloc + j) * 64 + t * 16 + n_in] = acc[t][j];
    __syncthreads();

    // 256 threads x 4 consecutive floats = the 16x64 context tile
    {
        int idx = threadIdx.x * 4;
        v4f ssum = v4f{0.f};
#pragma unroll
        for (int w = 0; w < 8; ++w) {
            v4f part = *(const v4f*)(smem + w * 1024 + idx);
            ssum += part;
        }
        int m = idx >> 6, d = idx & 63;
        *(v4f*)(attn_out + ((size_t)(b * SEQ + q0 + m)) * DMODEL + h * DHEAD + d) = ssum;
    }
}

// ---------------------------------------------------------------------------
// host launch
// ---------------------------------------------------------------------------
extern "C" void kernel_launch(void* const* d_in, const int* in_sizes, int n_in,
                              void* d_out, int out_size, void* d_ws, size_t ws_size,
                              hipStream_t stream) {
    const float* Q  = (const float*)d_in[0];
    const float* K  = (const float*)d_in[1];
    const float* V  = (const float*)d_in[2];
    const unsigned char* mask = (const unsigned char*)d_in[3];
    const float* Wq = (const float*)d_in[4];
    const float* bq = (const float*)d_in[5];
    const float* Wk = (const float*)d_in[6];
    const float* bk = (const float*)d_in[7];
    const float* Wv = (const float*)d_in[8];
    const float* bv = (const float*)d_in[9];
    const float* Wo = (const float*)d_in[10];
    const float* bo = (const float*)d_in[11];

    float* out        = (float*)d_out;
    float* weight_out = out + (size_t)MROWS * DMODEL;   // second tuple output

    // workspace layout (~42 MB)
    _Float16* Wq_t = (_Float16*)d_ws;
    _Float16* Wk_t = Wq_t + (size_t)DMODEL * DMODEL;
    _Float16* Wv_t = Wk_t + (size_t)DMODEL * DMODEL;
    _Float16* Wo_t = Wv_t + (size_t)DMODEL * DMODEL;
    _Float16* q_h  = Wo_t + (size_t)DMODEL * DMODEL;    // [8192,512] f16
    _Float16* k_h  = q_h  + (size_t)MROWS * DMODEL;
    _Float16* v_t  = k_h  + (size_t)MROWS * DMODEL;     // [B,H,64,S] f16
    float*  attn_f = (float*)(v_t + (size_t)MROWS * DMODEL);  // [8192,512] f32

    const int wblocks = (DMODEL * DMODEL) / 256;        // 1024
    wtrans_kernel<<<wblocks, 256, 0, stream>>>(Wq, Wq_t);
    wtrans_kernel<<<wblocks, 256, 0, stream>>>(Wk, Wk_t);
    wtrans_kernel<<<wblocks, 256, 0, stream>>>(Wv, Wv_t);
    wtrans_kernel<<<wblocks, 256, 0, stream>>>(Wo, Wo_t);

    dim3 ggrid(MROWS / 128, DMODEL / 64);               // (64, 8)
    gemm_kernel<<<ggrid, 256, 0, stream>>>(Q, Wq_t, bq, (void*)q_h, 0);
    gemm_kernel<<<ggrid, 256, 0, stream>>>(K, Wk_t, bk, (void*)k_h, 0);
    gemm_kernel<<<ggrid, 256, 0, stream>>>(V, Wv_t, bv, (void*)v_t, 1);

    attn_kernel<<<BB * NH * (SEQ / 16), 256, 0, stream>>>(q_h, k_h, v_t, mask,
                                                          weight_out, attn_f);

    gemm_kernel<<<ggrid, 256, 0, stream>>>(attn_f, Wo_t, bo, (void*)out, 2);
}
